// GraphNeuralNetwork_84817014161810
// MI455X (gfx1250) — matrix operand order repaired and checked by reading, not compiled, not run on hardware
//
#include <hip/hip_runtime.h>

typedef __attribute__((ext_vector_type(16))) _Float16 v16h;
typedef __attribute__((ext_vector_type(8)))  float    v8f;

#define NNODES 50000
#define NEDGES 400000
#define NBATCH 64
#define ETOT   (NEDGES + NNODES)
#define SLOPE  0.2f

// ---------- ordered-uint encoding for float atomic max ----------
__device__ __forceinline__ unsigned enc_f32(float f) {
  unsigned u = __float_as_uint(f);
  return (u & 0x80000000u) ? ~u : (u | 0x80000000u);
}
__device__ __forceinline__ float dec_f32(unsigned e) {
  unsigned u = (e & 0x80000000u) ? (e ^ 0x80000000u) : ~e;
  return __uint_as_float(u);
}
#define ENC_NEG_INF 0x007FFFFFu  // enc_f32(-inf)

// ---------- generic fill ----------
__global__ __launch_bounds__(256) void fill_u32(unsigned* __restrict__ p, unsigned v, int n) {
  int i = blockIdx.x * 256 + threadIdx.x;
  if (i < n) p[i] = v;
}

__device__ __forceinline__ void pack8(v16h& f, const int base, float4 a, float4 b) {
  f[base + 0] = (_Float16)a.x; f[base + 1] = (_Float16)a.y;
  f[base + 2] = (_Float16)a.z; f[base + 3] = (_Float16)a.w;
  f[base + 4] = (_Float16)b.x; f[base + 5] = (_Float16)b.y;
  f[base + 6] = (_Float16)b.z; f[base + 7] = (_Float16)b.w;
}

// ---------- WMMA GEMM: C[M,Nc] = A[M,K] * B[K,Nc], f16 inputs / f32 accum ----------
// M must be a multiple of 16 (true for 50000 and 64). Wave-per-16x16 tile,
// 8 row tiles per block; grid.x = Nc/16, grid.y = ceil(rowTiles/8).
// K and Nc are compile-time so all strided accesses use immediate offsets.
template <int K, int Nc>
__global__ __launch_bounds__(256) void wmma_gemm_f32(
    const float* __restrict__ A, const float* __restrict__ B,
    float* __restrict__ C, int numRowTiles) {
  const int wave = threadIdx.x >> 5;
  const int lane = threadIdx.x & 31;
  const int rowTile = blockIdx.y * 8 + wave;
  if (rowTile >= numRowTiles) return;           // uniform per wave
  const int laneM = lane & 15;
  const int hi = lane >> 4;                     // 0 or 1
  const int row = rowTile * 16 + laneM;
  const int col = blockIdx.x * 16 + laneM;
  const float* Arow = A + row * K;

  v8f acc = {};
#pragma unroll
  for (int k0 = 0; k0 < K; k0 += 32) {
    v16h af, bf;
    // A fragment (16x32, MxK): lane holds row M=lane%16.
    // halves 0..7  <- K = k0 + hi*8 + [0..7]
    // halves 8..15 <- K = k0 + hi*8 + [16..23]   (zero when K==16)
    const int ka = k0 + hi * 8;
    float4 a0 = *(const float4*)(Arow + ka);
    float4 a1 = *(const float4*)(Arow + ka + 4);
    pack8(af, 0, a0, a1);
    if constexpr (K >= 32) {
      float4 a2 = *(const float4*)(Arow + ka + 16);
      float4 a3 = *(const float4*)(Arow + ka + 20);
      pack8(af, 8, a2, a3);
    } else {
#pragma unroll
      for (int i = 8; i < 16; ++i) af[i] = (_Float16)0.f;
    }
    // B fragment (32x16, KxN): lane holds col N=lane%16; half i <- K = k0 + hi*8 + (i<8 ? i : i+8)
    if constexpr (K >= 32) {
      const float* Bp = B + (k0 + hi * 8) * Nc + col;
#pragma unroll
      for (int i = 0; i < 16; ++i) bf[i] = (_Float16)Bp[(i < 8 ? i : i + 8) * Nc];
    } else {
      // K==16: halves 8..15 would hold K=16..31 -> zeros
      {
        const float* Bp = B + hi * 8 * Nc + col;
#pragma unroll
        for (int i = 0; i < 8; ++i) bf[i] = (_Float16)Bp[i * Nc];
#pragma unroll
        for (int i = 8; i < 16; ++i) bf[i] = (_Float16)0.f;
      }
    }
    acc = __builtin_amdgcn_wmma_f32_16x16x32_f16(false, af, false, bf,
                                                 (short)0, acc, false, false);
  }
  // C/D layout: VGPR r -> row = rowTile*16 + hi*8 + r, col = lane%16
  float* Cp = C + (rowTile * 16 + hi * 8) * Nc + col;
#pragma unroll
  for (int r = 0; r < 8; ++r) Cp[r * Nc] = acc[r];
}

// ---------- attention logits: als/ald[n,h] = <hW[n,h,:], a_s/a_d[h,:]> ----------
__global__ __launch_bounds__(256) void attn_logits(
    const float* __restrict__ hW, const float* __restrict__ a_s,
    const float* __restrict__ a_d, float* __restrict__ als,
    float* __restrict__ ald, int H) {
  int t = blockIdx.x * 256 + threadIdx.x;
  if (t >= NNODES * H) return;
  int h = t % H;
  int n = t / H;
  const float* hv = hW + n * (H * 64) + h * 64;
  const float* as = a_s + h * 64;
  const float* ad = a_d + h * 64;
  float ss = 0.f, sd = 0.f;
#pragma unroll 4
  for (int c = 0; c < 64; ++c) { float v = hv[c]; ss += v * as[c]; sd += v * ad[c]; }
  als[t] = ss; ald[t] = sd;
}

// ---------- edge helpers ----------
__device__ __forceinline__ void edge_sd(const int* __restrict__ ei, int e,
                                        int& s, int& d) {
  if (e < NEDGES) { s = ei[e]; d = ei[NEDGES + e]; }
  else            { s = d = e - NEDGES; }   // self loops appended
}
__device__ __forceinline__ float leaky(float x) { return x > 0.f ? x : SLOPE * x; }

// pass 1: per-dst segment max of leaky logits
__global__ __launch_bounds__(256) void edge_max_k(
    const int* __restrict__ ei, const float* __restrict__ als,
    const float* __restrict__ ald, unsigned* __restrict__ mbits, int H) {
  int e = blockIdx.x * 256 + threadIdx.x;
  if (e >= ETOT) return;
  int s, d; edge_sd(ei, e, s, d);
  for (int h = 0; h < H; ++h) {
    float lg = leaky(als[s * H + h] + ald[d * H + h]);
    atomicMax(&mbits[d * H + h], enc_f32(lg));
  }
}

// pass 2: w = exp(e - m[dst]); z[dst] += w; stash w
__global__ __launch_bounds__(256) void edge_expsum_k(
    const int* __restrict__ ei, const float* __restrict__ als,
    const float* __restrict__ ald, const unsigned* __restrict__ mbits,
    float* __restrict__ wbuf, float* __restrict__ z, int H) {
  int t = blockIdx.x * 256 + threadIdx.x;
  if (t >= ETOT * H) return;
  int h = t % H;
  int e = t / H;
  int s, d; edge_sd(ei, e, s, d);
  float lg = leaky(als[s * H + h] + ald[d * H + h]);
  float mm = dec_f32(mbits[d * H + h]);
  float w = __expf(lg - mm);
  wbuf[t] = w;
  atomicAdd(&z[d * H + h], w);
}

// pass 3: acc[dst,h,c] += (w/z[dst,h]) * hW[src,h,c]
__global__ __launch_bounds__(256) void edge_agg_k(
    const int* __restrict__ ei, const float* __restrict__ wbuf,
    const float* __restrict__ z, const float* __restrict__ hW,
    float* __restrict__ acc, int H) {
  int t = blockIdx.x * 256 + threadIdx.x;          // < ETOT*H*64 = 115.2M
  if (t >= ETOT * H * 64) return;
  int c = t & 63;
  int eh = t >> 6;                                  // e*H + h
  int h = eh % H;
  int e = eh / H;
  int s, d; edge_sd(ei, e, s, d);
  float alpha = wbuf[eh] / z[d * H + h];
  float v = alpha * hW[s * (H * 64) + h * 64 + c];
  atomicAdd(&acc[d * (H * 64) + h * 64 + c], v);
}

// head mean + bias (+relu)
__global__ __launch_bounds__(256) void head_mean_k(
    const float* __restrict__ acc, const float* __restrict__ bias,
    float* __restrict__ out, int H, int doRelu) {
  int t = blockIdx.x * 256 + threadIdx.x;
  if (t >= NNODES * 64) return;
  int c = t & 63;
  int n = t >> 6;
  float s = 0.f;
  for (int h = 0; h < H; ++h) s += acc[n * (H * 64) + h * 64 + c];
  s = s / (float)H + bias[c];
  if (doRelu) s = fmaxf(s, 0.f);
  out[t] = s;
}

// ---------- pooling ----------
__global__ __launch_bounds__(256) void pool_accum_k(
    const float* __restrict__ hnode, const int* __restrict__ batch,
    float* __restrict__ meanacc, unsigned* __restrict__ maxbits,
    float* __restrict__ counts) {
  int t = blockIdx.x * 256 + threadIdx.x;
  if (t >= NNODES * 64) return;
  int c = t & 63;
  int n = t >> 6;
  int b = batch[n];
  float v = hnode[t];
  atomicAdd(&meanacc[b * 64 + c], v);
  atomicMax(&maxbits[b * 64 + c], enc_f32(v));
  if (c == 0) atomicAdd(&counts[b], 1.0f);
}

__global__ __launch_bounds__(256) void pool_final_k(
    const float* __restrict__ meanacc, const unsigned* __restrict__ maxbits,
    const float* __restrict__ counts, float* __restrict__ g) {
  int t = blockIdx.x * 256 + threadIdx.x;
  if (t >= NBATCH * 64) return;
  int c = t & 63; int b = t >> 6;
  g[b * 128 + c]      = meanacc[t] / counts[b];
  g[b * 128 + 64 + c] = dec_f32(maxbits[t]);
}

__global__ __launch_bounds__(256) void add_bias_k(float* __restrict__ out,
                                                  const float* __restrict__ bp) {
  int t = blockIdx.x * 256 + threadIdx.x;
  if (t >= NBATCH * 64) return;
  out[t] += bp[t & 63];
}

// ---------------------------------------------------------------------------
extern "C" void kernel_launch(void* const* d_in, const int* in_sizes, int n_in,
                              void* d_out, int out_size, void* d_ws, size_t ws_size,
                              hipStream_t stream) {
  const float* X    = (const float*)d_in[0];
  const int*   ei   = (const int*)d_in[1];
  const int*   batch= (const int*)d_in[2];
  const float* W1   = (const float*)d_in[3];
  const float* a1s  = (const float*)d_in[4];
  const float* a1d  = (const float*)d_in[5];
  const float* b1   = (const float*)d_in[6];
  const float* W2   = (const float*)d_in[7];
  const float* a2s  = (const float*)d_in[8];
  const float* a2d  = (const float*)d_in[9];
  const float* b2   = (const float*)d_in[10];
  const float* W3   = (const float*)d_in[11];
  const float* a3s  = (const float*)d_in[12];
  const float* a3d  = (const float*)d_in[13];
  const float* b3   = (const float*)d_in[14];
  const float* Wp   = (const float*)d_in[15];
  const float* bp   = (const float*)d_in[16];
  float* out = (float*)d_out;

  // workspace carve-up (floats)
  float* ws = (float*)d_ws;
  size_t off = 0;
  float*    hW      = ws + off; off += (size_t)NNODES * 256;
  float*    acc     = ws + off; off += (size_t)NNODES * 256;
  float*    hnode   = ws + off; off += (size_t)NNODES * 64;
  float*    als     = ws + off; off += (size_t)NNODES * 4;
  float*    ald     = ws + off; off += (size_t)NNODES * 4;
  unsigned* mbits   = (unsigned*)(ws + off); off += (size_t)NNODES * 4;
  float*    z       = ws + off; off += (size_t)NNODES * 4;
  float*    wbuf    = ws + off; off += (size_t)ETOT * 4;
  float*    counts  = ws + off; off += NBATCH;
  float*    meanacc = ws + off; off += NBATCH * 64;
  unsigned* maxbits = (unsigned*)(ws + off); off += NBATCH * 64;
  float*    g       = ws + off; off += NBATCH * 128;

  auto blocks = [](long long n) { return (unsigned)((n + 255) / 256); };

  const int RT = NNODES / 16;              // 3125 row tiles, exact
  const dim3 gridN256(256 / 16, (RT + 7) / 8);
  const dim3 gridN64 (64  / 16, (RT + 7) / 8);

  // ---- per-layer pipeline (edge-softmax part shared) ----
  auto edge_part = [&](const float* as_, const float* ad_, const float* bias,
                       int H, int doRelu, float* hout) {
    const int HC = H * 64;
    int nh = NNODES * H;
    attn_logits<<<blocks(nh), 256, 0, stream>>>(hW, as_, ad_, als, ald, H);
    fill_u32<<<blocks(nh), 256, 0, stream>>>(mbits, ENC_NEG_INF, nh);
    fill_u32<<<blocks(nh), 256, 0, stream>>>((unsigned*)z, 0u, nh);
    int accn = NNODES * HC;
    fill_u32<<<blocks(accn), 256, 0, stream>>>((unsigned*)acc, 0u, accn);
    edge_max_k<<<blocks(ETOT), 256, 0, stream>>>(ei, als, ald, mbits, H);
    int eh = ETOT * H;
    edge_expsum_k<<<blocks(eh), 256, 0, stream>>>(ei, als, ald, mbits, wbuf, z, H);
    edge_agg_k<<<blocks((long long)eh * 64), 256, 0, stream>>>(ei, wbuf, z, hW, acc, H);
    head_mean_k<<<blocks(NNODES * 64), 256, 0, stream>>>(acc, bias, hout, H, doRelu);
  };

  // layer 1: GATConv(16 -> 64, heads=4)
  wmma_gemm_f32<16, 256><<<gridN256, 256, 0, stream>>>(X, W1, hW, RT);
  edge_part(a1s, a1d, b1, 4, 1, hnode);
  // layer 2: GATConv(64 -> 64, heads=4)
  wmma_gemm_f32<64, 256><<<gridN256, 256, 0, stream>>>(hnode, W2, hW, RT);
  edge_part(a2s, a2d, b2, 4, 1, hnode);
  // layer 3: GATConv(64 -> 64, heads=1)
  wmma_gemm_f32<64, 64><<<gridN64, 256, 0, stream>>>(hnode, W3, hW, RT);
  edge_part(a3s, a3d, b3, 1, 0, hnode);

  // pooling
  fill_u32<<<1, 256, 0, stream>>>((unsigned*)counts, 0u, NBATCH);
  fill_u32<<<blocks(NBATCH * 64), 256, 0, stream>>>((unsigned*)meanacc, 0u, NBATCH * 64);
  fill_u32<<<blocks(NBATCH * 64), 256, 0, stream>>>(maxbits, ENC_NEG_INF, NBATCH * 64);
  pool_accum_k<<<blocks(NNODES * 64), 256, 0, stream>>>(hnode, batch, meanacc, maxbits, counts);
  pool_final_k<<<blocks(NBATCH * 64), 256, 0, stream>>>(meanacc, maxbits, counts, g);

  // final projection: out = g @ Wp + bp  (WMMA, M=64, K=128, N=64)
  dim3 gridp(64 / 16, 1);
  wmma_gemm_f32<128, 64><<<gridp, 256, 0, stream>>>(g, Wp, out, NBATCH / 16);
  add_bias_k<<<blocks(NBATCH * 64), 256, 0, stream>>>(out, bp);
}
